// MLPgc_6485400617149
// MI455X (gfx1250) — matrix operand order repaired and checked by reading, not compiled, not run on hardware
//
#include <hip/hip_runtime.h>
#include <hip/hip_bf16.h>

typedef __attribute__((ext_vector_type(16))) __bf16 v16bf;
typedef __attribute__((ext_vector_type(8)))  __bf16 v8bf;
typedef __attribute__((ext_vector_type(8)))  float  v8f;
typedef int v4i __attribute__((vector_size(16)));   // matches builtin param type

union V16U { v16bf v; v8bf h[2]; };

#define BATCH 16384
#define NVARS 64
#define LAG   16
#define HD1   64
#define HD2   1024
#define KDIM  4096   // NVARS*HD1

#if __has_builtin(__builtin_amdgcn_global_load_async_to_lds_b128) && \
    __has_builtin(__builtin_amdgcn_s_wait_asynccnt)
#define USE_ASYNC_LDS 1
#else
#define USE_ASYNC_LDS 0
#endif

#if USE_ASYNC_LDS
#define ASYNC_B128(g, l)                                                        \
  __builtin_amdgcn_global_load_async_to_lds_b128(                               \
      (__attribute__((address_space(1))) v4i*)(g),                              \
      (__attribute__((address_space(3))) v4i*)(l), 0, 0)
#else
#define ASYNC_B128(g, l) ((void)0)
#endif

// A-fragment (16x32, bf16) from row-major LDS [row][k], pitch `stride` elems.
// lanes 0-15 / 16-31 hold M = lane%16; elems 0..7 -> K = hi*8..+7,
// elems 8..15 -> K = 16+hi*8..+7.  Two 16B ds_load_b128 per lane.
static __device__ __forceinline__ v16bf load_a_frag(const __bf16* buf, int stride,
                                                    int row0, int k0, int lane) {
  const int m  = lane & 15;
  const int hi = lane >> 4;
  const __bf16* p = buf + (row0 + m) * stride + k0 + hi * 8;
  V16U u;
  u.h[0] = *(const v8bf*)(p);
  u.h[1] = *(const v8bf*)(p + 16);
  return u.v;
}

// B-fragment (32x16, bf16) from N-major LDS [n][k], pitch `stride` elems.
// lane%16 = N; elems 0..7 -> K = hi*8..+7, elems 8..15 -> K = 16+hi*8..+7.
static __device__ __forceinline__ v16bf load_b_frag(const __bf16* buf, int stride,
                                                    int n0, int k0, int lane) {
  const int n  = n0 + (lane & 15);
  const int hi = lane >> 4;
  const __bf16* p = buf + n * stride + k0 + hi * 8;
  V16U u;
  u.h[0] = *(const v8bf*)(p);
  u.h[1] = *(const v8bf*)(p + 16);
  return u.v;
}

// ------------------- W3 [4096,1024] f32  ->  w3t [1024,4096] bf16 (tiled) ----
// grid: (HD2/64, KDIM/64); block 256
__global__ __launch_bounds__(256) void k_transpose_w3(const float* __restrict__ W3,
                                                      __bf16* __restrict__ w3t) {
  __shared__ __bf16 ts[64 * 72];
  const int tid = threadIdx.x;
  const int N0 = blockIdx.x * 64;
  const int K0 = blockIdx.y * 64;
#pragma unroll
  for (int i = 0; i < 16; ++i) {          // 4096 elements, coalesced f32 reads
    int idx = tid + i * 256;
    int kl = idx >> 6, nl = idx & 63;
    ts[nl * 72 + kl] = (__bf16)W3[(size_t)(K0 + kl) * HD2 + N0 + nl];
  }
  __syncthreads();
#pragma unroll
  for (int i = 0; i < 2; ++i) {           // 512 chunks of 8 -> 16B stores
    int chunk = tid + i * 256;
    int nl = chunk >> 3, kc = (chunk & 7) * 8;
    *(v8bf*)(w3t + (size_t)(N0 + nl) * KDIM + K0 + kc) =
        *(const v8bf*)(ts + nl * 72 + kc);
  }
}

// ---------------------------- fused per-variable layers 1+2 (WMMA bf16) ----
// grid: (BATCH/128, NVARS); block: 256 (8 waves, 16 rows per wave)
__global__ __launch_bounds__(256) void k_fused12(
    const float* __restrict__ inputs, const float* __restrict__ W1,
    const float* __restrict__ b1, const float* __restrict__ W2,
    const float* __restrict__ b2, const float* __restrict__ imp,
    __bf16* __restrict__ agg) {
  __shared__ __bf16 xs [128 * 40];   // x tile, K padded 16->32, pitch 40
  __shared__ __bf16 w1t[ 64 * 40];   // W1^T [n][k], K padded 16->32
  __shared__ __bf16 w2t[ 64 * 72];   // W2^T [n][k], K=64, pitch 72
  __shared__ __bf16 h1s[128 * 72];   // h1 tile (reused for output staging)
  __shared__ float  b1s[64], b2s[64];

  const int tid = threadIdx.x;
  const int v   = blockIdx.y;
  const int M0  = blockIdx.x * 128;

#pragma unroll
  for (int i = 0; i < 10; ++i) {               // 2560 = 64*40
    int idx = tid + i * 256;
    int n = idx / 40, k = idx % 40;
    float val = (k < LAG) ? W1[(size_t)v * (LAG * HD1) + k * HD1 + n] : 0.0f;
    w1t[idx] = (__bf16)val;
  }
#pragma unroll
  for (int i = 0; i < 18; ++i) {               // 4608 = 64*72
    int idx = tid + i * 256;
    int n = idx / 72, k = idx % 72;
    float val = (k < HD1) ? W2[(size_t)v * (HD1 * HD1) + k * HD1 + n] : 0.0f;
    w2t[idx] = (__bf16)val;
  }
#pragma unroll
  for (int i = 0; i < 20; ++i) {               // 5120 = 128*40
    int idx = tid + i * 256;
    int r = idx / 40, k = idx % 40;
    float val = (k < LAG) ? inputs[(size_t)(M0 + r) * 1024 + v * LAG + k] : 0.0f;
    xs[idx] = (__bf16)val;
  }
  if (tid < 64) { b1s[tid] = b1[v * HD1 + tid]; b2s[tid] = b2[v * HD1 + tid]; }
  const float iv = imp[v];
  __syncthreads();

  const int wave = tid >> 5;
  const int lane = tid & 31;
  const int hi   = lane >> 4;
  const int nl   = lane & 15;
  const int R0   = wave * 16;

  // ---- layer 1: h1 = relu(x @ W1 + b1), K = 32 (zero padded) ----
  v16bf a = load_a_frag(xs, 40, R0, 0, lane);
#pragma unroll
  for (int nt = 0; nt < 4; ++nt) {
    v16bf bm = load_b_frag(w1t, 40, nt * 16, 0, lane);
    v8f c = {};
    c = __builtin_amdgcn_wmma_f32_16x16x32_bf16(false, a, false, bm, (short)0, c,
                                                false, false);
    float bias = b1s[nt * 16 + nl];
#pragma unroll
    for (int r = 0; r < 8; ++r) {
      float h = c[r] + bias;
      h = h > 0.0f ? h : 0.0f;
      h1s[(R0 + r + hi * 8) * 72 + nt * 16 + nl] = (__bf16)h;
    }
  }
  // Same-wave LDS RAW: DS ops are in-order within a wave -> no barrier needed.

  // ---- layer 2: h2 = relu(h1 @ W2 + b2) * imp[v], K = 64 ----
  v16bf a0 = load_a_frag(h1s, 72, R0, 0, lane);
  v16bf a1 = load_a_frag(h1s, 72, R0, 32, lane);
#pragma unroll
  for (int nt = 0; nt < 4; ++nt) {
    v8f c = {};
    c = __builtin_amdgcn_wmma_f32_16x16x32_bf16(
        false, a0, false, load_b_frag(w2t, 72, nt * 16, 0, lane), (short)0, c,
        false, false);
    c = __builtin_amdgcn_wmma_f32_16x16x32_bf16(
        false, a1, false, load_b_frag(w2t, 72, nt * 16, 32, lane), (short)0, c,
        false, false);
    float bias = b2s[nt * 16 + nl];
#pragma unroll
    for (int r = 0; r < 8; ++r) {
      float h = c[r] + bias;
      h = h > 0.0f ? h : 0.0f;
      // stage result back into h1s (own rows; a0/a1 already in registers)
      h1s[(R0 + r + hi * 8) * 72 + nt * 16 + nl] = (__bf16)(h * iv);
    }
  }
  __syncthreads();
  // coalesced 16B stores of the 128x64 output tile
#pragma unroll
  for (int i = 0; i < 4; ++i) {                // 1024 chunks of 8
    int chunk = tid + i * 256;
    int r = chunk >> 3, cb = (chunk & 7) * 8;
    *(v8bf*)(agg + (size_t)(M0 + r) * KDIM + v * HD1 + cb) =
        *(const v8bf*)(h1s + r * 72 + cb);
  }
}

// ----------------- layer 3: h3 = relu(agg @ W3 + b3)  (128x128 WMMA tiles) --
// grid: (HD2/128, BATCH/128); block 256 = 8 waves in 2(M)x4(N) arrangement
// Double-buffered LDS; async global->LDS copies when the toolchain has them.
__global__ __launch_bounds__(256) void k_gemm3(const __bf16* __restrict__ agg,
                                               const __bf16* __restrict__ w3t,
                                               const float* __restrict__ b3,
                                               __bf16* __restrict__ h3) {
  // 40KB: two (A 5120 + B 5120) tile buffers; reused as 128x136 out staging
  __shared__ __bf16 smem[20480];
  __bf16* const Ab[2] = { smem,        smem + 10240 };
  __bf16* const Bb[2] = { smem + 5120, smem + 15360 };

  const int tid  = threadIdx.x;
  const int N0   = blockIdx.x * 128;
  const int M0   = blockIdx.y * 128;
  const int wave = tid >> 5;
  const int lane = tid & 31;
  const int hi   = lane >> 4;
  const int nl   = lane & 15;
  const int RM   = (wave & 1) * 64;   // wave's 64-row stripe
  const int CN   = (wave >> 1) * 32;  // wave's 32-col stripe

  const int r  = tid >> 2;            // 0..63 (i=0), +64 (i=1)
  const int c4 = (tid & 3) * 8;       // 8-elem chunk within the 32-wide K step

  v8f c[4][2] = {};

#define LOAD_TILES(K0, buf)                                                     \
  do {                                                                          \
    _Pragma("unroll") for (int i = 0; i < 2; ++i) {                             \
      int rr = r + i * 64;                                                      \
      const __bf16* ga = agg + (size_t)(M0 + rr) * KDIM + (K0) + c4;            \
      const __bf16* gb = w3t + (size_t)(N0 + rr) * KDIM + (K0) + c4;            \
      __bf16* la = Ab[buf] + rr * 40 + c4;                                      \
      __bf16* lb = Bb[buf] + rr * 40 + c4;                                      \
      if (USE_ASYNC_LDS) {                                                      \
        ASYNC_B128(ga, la);                                                     \
        ASYNC_B128(gb, lb);                                                     \
      } else {                                                                  \
        *(v8bf*)la = *(const v8bf*)ga;                                          \
        *(v8bf*)lb = *(const v8bf*)gb;                                          \
      }                                                                         \
    }                                                                           \
  } while (0)

  LOAD_TILES(0, 0);
  for (int ks = 0; ks < KDIM / 32; ++ks) {
    const int cur = ks & 1;
    if (ks + 1 < KDIM / 32) {
      LOAD_TILES((ks + 1) * 32, cur ^ 1);
#if USE_ASYNC_LDS
      __builtin_amdgcn_s_wait_asynccnt(4);   // this wave's current tiles done
#endif
    } else {
#if USE_ASYNC_LDS
      __builtin_amdgcn_s_wait_asynccnt(0);
#endif
    }
    __syncthreads();

    const __bf16* Ac = Ab[cur];
    const __bf16* Bc = Bb[cur];
    v16bf bf0 = load_b_frag(Bc, 40, CN, 0, lane);
    v16bf bf1 = load_b_frag(Bc, 40, CN + 16, 0, lane);
#pragma unroll
    for (int mt = 0; mt < 4; ++mt) {
      v16bf af = load_a_frag(Ac, 40, RM + mt * 16, 0, lane);
      c[mt][0] = __builtin_amdgcn_wmma_f32_16x16x32_bf16(false, af, false, bf0,
                                                         (short)0, c[mt][0],
                                                         false, false);
      c[mt][1] = __builtin_amdgcn_wmma_f32_16x16x32_bf16(false, af, false, bf1,
                                                         (short)0, c[mt][1],
                                                         false, false);
    }
    __syncthreads();
  }

  // epilogue: bias + relu, stage 128x136 tile in LDS, coalesced 16B stores
#pragma unroll
  for (int nt = 0; nt < 2; ++nt) {
    float bias = b3[N0 + CN + nt * 16 + nl];
#pragma unroll
    for (int mt = 0; mt < 4; ++mt) {
#pragma unroll
      for (int rr = 0; rr < 8; ++rr) {
        float hv = c[mt][nt][rr] + bias;
        hv = hv > 0.0f ? hv : 0.0f;
        smem[(RM + mt * 16 + rr + hi * 8) * 136 + CN + nt * 16 + nl] = (__bf16)hv;
      }
    }
  }
  __syncthreads();
#pragma unroll
  for (int i = 0; i < 8; ++i) {               // 2048 chunks of 8
    int chunk = tid + i * 256;
    int rr = chunk >> 4, cb = (chunk & 15) * 8;
    *(v8bf*)(h3 + (size_t)(M0 + rr) * HD2 + N0 + cb) =
        *(const v8bf*)(smem + rr * 136 + cb);
  }
}

// ------------------------------- layer 4: out = h3 @ W4 + b4 (one wave/row) -
__global__ __launch_bounds__(256) void k_l4(const __bf16* __restrict__ h3,
                                            const float* __restrict__ W4,
                                            const float* __restrict__ b4,
                                            float* __restrict__ out) {
  const int wave = threadIdx.x >> 5;
  const int lane = threadIdx.x & 31;
  const int m = blockIdx.x * 8 + wave;
  const __bf16* row = h3 + (size_t)m * HD2;
  float sum = 0.0f;
#pragma unroll
  for (int it = 0; it < 4; ++it) {
    int base = (it * 32 + lane) * 8;
    v8bf h = *(const v8bf*)(row + base);
    float4 w0 = *(const float4*)(W4 + base);
    float4 w1 = *(const float4*)(W4 + base + 4);
    sum += (float)h[0] * w0.x + (float)h[1] * w0.y + (float)h[2] * w0.z +
           (float)h[3] * w0.w + (float)h[4] * w1.x + (float)h[5] * w1.y +
           (float)h[6] * w1.z + (float)h[7] * w1.w;
  }
#pragma unroll
  for (int off = 16; off > 0; off >>= 1) sum += __shfl_down(sum, off);
  if (lane == 0) out[m] = sum + b4[0];
}

extern "C" void kernel_launch(void* const* d_in, const int* in_sizes, int n_in,
                              void* d_out, int out_size, void* d_ws, size_t ws_size,
                              hipStream_t stream) {
  const float* inputs = (const float*)d_in[0];
  const float* W1 = (const float*)d_in[1];
  const float* b1 = (const float*)d_in[2];
  const float* W2 = (const float*)d_in[3];
  const float* b2 = (const float*)d_in[4];
  const float* imp = (const float*)d_in[5];
  const float* W3 = (const float*)d_in[6];
  const float* b3 = (const float*)d_in[7];
  const float* W4 = (const float*)d_in[8];
  const float* b4 = (const float*)d_in[9];
  float* out = (float*)d_out;

  // workspace (bf16): agg [B,4096] | W3^T [1024,4096] | h3 [B,1024]
  __bf16* agg = (__bf16*)d_ws;
  __bf16* w3t = agg + (size_t)BATCH * KDIM;
  __bf16* h3  = w3t + (size_t)HD2 * KDIM;

  k_transpose_w3<<<dim3(HD2 / 64, KDIM / 64), 256, 0, stream>>>(W3, w3t);
  k_fused12<<<dim3(BATCH / 128, NVARS), 256, 0, stream>>>(inputs, W1, b1, W2, b2,
                                                          imp, agg);
  k_gemm3<<<dim3(HD2 / 128, BATCH / 128), 256, 0, stream>>>(agg, w3t, b3, h3);
  k_l4<<<BATCH / 8, 256, 0, stream>>>(h3, W4, b4, out);

  (void)in_sizes; (void)n_in; (void)out_size; (void)ws_size;
}
